// MultiHeadQGFDLayer_64183991271491
// MI455X (gfx1250) — matrix operand order repaired and checked by reading, not compiled, not run on hardware
//
#include <hip/hip_runtime.h>
#include <hip/hip_bf16.h>

// ---------------- problem constants (from reference) ----------------
#define BATCH   2
#define L_SEQ   1536
#define DMODEL  1024
#define NHEAD   16
#define PDIM    1024
#define HDIM    64          // PDIM / NHEAD
#define MROWS   (BATCH * L_SEQ)   // 3072
#define KSIZE   5
#define DIFFST  3
#define ALPHA   0.02f
#define NEGV    (-1.0e9f)
#define EPSV    1e-9f
#define SPLD    1552        // padded LDS row stride (halves), 1536 + 16

typedef __attribute__((ext_vector_type(16))) _Float16 v16h;
typedef __attribute__((ext_vector_type(8)))  float    v8f;
typedef __attribute__((vector_size(16)))     int      v4i;   // builtin's pointee type

#define AS_GLOBAL __attribute__((address_space(1)))
#define AS_LDS    __attribute__((address_space(3)))

// CDNA5 async global->LDS copy (ASYNCcnt-tracked), compile-safe detection.
#if defined(__gfx1250__) && \
    __has_builtin(__builtin_amdgcn_global_load_async_to_lds_b128) && \
    __has_builtin(__builtin_amdgcn_s_wait_asynccnt)
#define HAVE_ASYNC_LDS 1
#else
#define HAVE_ASYNC_LDS 0
#endif

union F16Frag {
    v16h v;
    _Float16 h[16];
    uint4 q[2];
};

__device__ __forceinline__ v8f wmma_f16(const v16h& a, const v16h& b, const v8f& c) {
    // D = A(16x32 f16) x B(32x16 f16) + C(16x16 f32)
    return __builtin_amdgcn_wmma_f32_16x16x32_f16(
        /*neg_a=*/false, a, /*neg_b=*/false, b,
        /*c_mod=*/(short)0, c, /*reuse_a=*/false, /*reuse_b=*/false);
}

__device__ __forceinline__ void async_wait0() {
#if HAVE_ASYNC_LDS
    __builtin_amdgcn_s_wait_asynccnt(0);   // s_wait_asynccnt 0
#endif
}

// ---------------- helper kernels ----------------

__global__ __launch_bounds__(256)
void k_cvt_f16(const float* __restrict__ src, _Float16* __restrict__ dst, int n) {
    int i = blockIdx.x * 256 + threadIdx.x;
    if (i < n) dst[i] = (_Float16)src[i];
}

// src: f32 row-major [K][N]  ->  dst: f16 [N][K] (transposed, K contiguous)
__global__ __launch_bounds__(256)
void k_cvt_transpose(const float* __restrict__ src, _Float16* __restrict__ dst,
                     int K, int N) {
    int n = blockIdx.x * 32 + (threadIdx.x & 31);
    int k = blockIdx.y * 8 + (threadIdx.x >> 5);
    if (n < N && k < K)
        dst[(size_t)n * K + k] = (_Float16)src[(size_t)k * N + n];
}

// V: f16 [B*L][P]  ->  Vt: f16 [B][H][HD][L]
__global__ __launch_bounds__(128)
void k_transpose_v(const _Float16* __restrict__ V, _Float16* __restrict__ Vt) {
    int l  = blockIdx.x * 128 + threadIdx.x;   // 0..L-1
    int d  = blockIdx.y;                       // 0..HD-1
    int bh = blockIdx.z;                       // 0..B*H-1
    int b = bh / NHEAD, h = bh % NHEAD;
    Vt[((size_t)bh * HDIM + d) * L_SEQ + l] =
        V[((size_t)b * L_SEQ + l) * PDIM + h * HDIM + d];
}

// ---------------- WMMA GEMM: out = A(f16,[M][K]) * Bt(f16,[N][K])^T + bias ----------------
// block = 128 threads (4 waves); block tile M=32, N=128; each wave: 32x32 (4 WMMA tiles).
// A tile (32x32 halves, shared by all 4 waves) is double-buffered in LDS via
// GLOBAL_LOAD_ASYNC_TO_LDS_B128 (ASYNCcnt) when available.
__global__ __launch_bounds__(128)
void k_gemm(const _Float16* __restrict__ A, int lda,
            const _Float16* __restrict__ Bt, int ldb,
            const float* __restrict__ bias,
            void* __restrict__ outp, int ldo,
            int K, int out_f32) {
    __shared__ _Float16 ldsA[2][32 * 32];      // 2 x 2 KB double buffer

    const int tid  = threadIdx.x;
    const int wave = tid >> 5;
    const int lane = tid & 31;
    const int ln   = lane & 15;
    const int hi   = lane >> 4;

    const int m0 = blockIdx.x * 32;
    const int n0 = blockIdx.y * 128 + wave * 32;

    v8f acc00 = {}, acc01 = {}, acc10 = {}, acc11 = {};

    const _Float16* B0 = Bt + (size_t)(n0 + ln) * ldb + 16 * hi;
    const _Float16* B1 = Bt + (size_t)(n0 + 16 + ln) * ldb + 16 * hi;

    // cooperative A-tile stage: 128 threads x 16B = 32 rows x 32 halves
    const int srow = tid >> 2;            // 0..31
    const int sseg = (tid & 3) * 8;       // 0,8,16,24 halves
    auto stage = [&](int k0, int bufi) {
        const _Float16* g = A + (size_t)(m0 + srow) * lda + k0 + sseg;
        _Float16* l = &ldsA[bufi][srow * 32 + sseg];
#if HAVE_ASYNC_LDS
        __builtin_amdgcn_global_load_async_to_lds_b128(
            (AS_GLOBAL v4i*)g, (AS_LDS v4i*)l, 0, 0);
#else
        *(uint4*)l = *(const uint4*)g;
#endif
    };

    stage(0, 0);
    int buf = 0;
    for (int k0 = 0; k0 < K; k0 += 32) {
        async_wait0();
        __syncthreads();
        if (k0 + 32 < K) stage(k0 + 32, buf ^ 1);   // overlap copy of next tile

        __builtin_prefetch(B0 + k0 + 128, 0, 0);    // global_prefetch_b8
        __builtin_prefetch(B1 + k0 + 128, 0, 0);

        const _Float16* la = &ldsA[buf][0];
        F16Frag a0, a1;                              // A rows m0..m0+15 / m0+16..m0+31
        a0.q[0] = *(const uint4*)(la + ln * 32 + 16 * hi);        // ds_load_b128
        a0.q[1] = *(const uint4*)(la + ln * 32 + 16 * hi + 8);
        a1.q[0] = *(const uint4*)(la + (16 + ln) * 32 + 16 * hi);
        a1.q[1] = *(const uint4*)(la + (16 + ln) * 32 + 16 * hi + 8);

        F16Frag b0, b1;
        b0.q[0] = *(const uint4*)(B0 + k0);
        b0.q[1] = *(const uint4*)(B0 + k0 + 8);
        b1.q[0] = *(const uint4*)(B1 + k0);
        b1.q[1] = *(const uint4*)(B1 + k0 + 8);

        acc00 = wmma_f16(a0.v, b0.v, acc00);
        acc01 = wmma_f16(a0.v, b1.v, acc01);
        acc10 = wmma_f16(a1.v, b0.v, acc10);
        acc11 = wmma_f16(a1.v, b1.v, acc11);
        buf ^= 1;
    }

    // C/D layout: lane holds n = ln, rows m = r + 8*hi
    const int nA = n0 + ln;
    const int nB = n0 + 16 + ln;
    const float biasA = bias[nA];
    const float biasB = bias[nB];
    for (int r = 0; r < 8; ++r) {
        int mA = m0 + r + 8 * hi;          // m-tile 0
        int mB = m0 + 16 + r + 8 * hi;     // m-tile 1
        float v00 = acc00[r] + biasA, v01 = acc01[r] + biasB;
        float v10 = acc10[r] + biasA, v11 = acc11[r] + biasB;
        if (out_f32) {
            float* o = (float*)outp;
            o[(size_t)mA * ldo + nA] = v00;
            o[(size_t)mA * ldo + nB] = v01;
            o[(size_t)mB * ldo + nA] = v10;
            o[(size_t)mB * ldo + nB] = v11;
        } else {
            _Float16* o = (_Float16*)outp;
            o[(size_t)mA * ldo + nA] = (_Float16)v00;
            o[(size_t)mA * ldo + nB] = (_Float16)v01;
            o[(size_t)mB * ldo + nA] = (_Float16)v10;
            o[(size_t)mB * ldo + nB] = (_Float16)v11;
        }
    }
}

// ---------------- LDS tree reductions (128 threads) ----------------
__device__ __forceinline__ float block_reduce_max(float v, float* red, int tid) {
    red[tid] = v; __syncthreads();
    for (int s = 64; s > 0; s >>= 1) {
        if (tid < s) red[tid] = fmaxf(red[tid], red[tid + s]);
        __syncthreads();
    }
    float r = red[0]; __syncthreads();
    return r;
}
__device__ __forceinline__ float block_reduce_sum(float v, float* red, int tid) {
    red[tid] = v; __syncthreads();
    for (int s = 64; s > 0; s >>= 1) {
        if (tid < s) red[tid] = red[tid] + red[tid + s];
        __syncthreads();
    }
    float r = red[0]; __syncthreads();
    return r;
}

// ---------------- fused attention: scores -> softmax -> 3x diffusion -> p@V ----------------
// grid (L/16, H, B), 128 threads (4 waves). Probability rows live in LDS (f16),
// never touch HBM.
__global__ __launch_bounds__(128)
void k_attention(const _Float16* __restrict__ Q,
                 const _Float16* __restrict__ Kb,
                 const _Float16* __restrict__ Vt,
                 const int* __restrict__ amask,
                 const float* __restrict__ kernel_w,
                 _Float16* __restrict__ attn) {
    __shared__ _Float16 sp[16 * SPLD];     // 49.7 KB: 16 prob rows (f16)
    __shared__ float maskf[L_SEQ];         // key-valid (1/0)
    __shared__ float red[128];
    __shared__ float wk[8];                // [0..4] smoothed taps

    const int tid  = threadIdx.x;
    const int wave = tid >> 5;
    const int lane = tid & 31;
    const int ln   = lane & 15;
    const int hi   = lane >> 4;

    const int q0 = blockIdx.x * 16;
    const int h  = blockIdx.y;
    const int b  = blockIdx.z;

    if (tid == 0) {
        float m = kernel_w[0];
        for (int j = 1; j < KSIZE; ++j) m = fmaxf(m, kernel_w[j]);
        float e[KSIZE], s = 0.f;
        for (int j = 0; j < KSIZE; ++j) { e[j] = __expf(kernel_w[j] - m); s += e[j]; }
        for (int j = 0; j < KSIZE; ++j) wk[j] = e[j] / s;
    }
    for (int k = tid; k < L_SEQ; k += 128)
        maskf[k] = (amask[b * L_SEQ + k] > 0) ? 1.f : 0.f;
    __syncthreads();

    // ---- phase 1: S = Q K^T / sqrt(HD) (per 16x16 key tile, WMMA) ----
    const _Float16* Qb = Q  + (size_t)(b * L_SEQ) * PDIM + h * HDIM;
    const _Float16* Kh = Kb + (size_t)(b * L_SEQ) * PDIM + h * HDIM;
    F16Frag aq0, aq1;   // A fragments for d0 = 0 and d0 = 32 (K dim = HDIM = 64)
    {
        const _Float16* qr = Qb + (size_t)(q0 + ln) * PDIM;
        aq0.q[0] = *(const uint4*)(qr + 8 * hi);
        aq0.q[1] = *(const uint4*)(qr + 16 + 8 * hi);
        aq1.q[0] = *(const uint4*)(qr + 32 + 8 * hi);
        aq1.q[1] = *(const uint4*)(qr + 48 + 8 * hi);
    }
    const float inv_sqrt_hd = 0.125f;
    for (int nt = wave; nt < L_SEQ / 16; nt += 4) {
        const int key0 = nt * 16;
        const _Float16* kr = Kh + (size_t)(key0 + ln) * PDIM + 8 * hi;
        F16Frag b0, b1;                     // B: K^T, columns = key positions
        b0.q[0] = *(const uint4*)(kr);
        b0.q[1] = *(const uint4*)(kr + 16);
        b1.q[0] = *(const uint4*)(kr + 32);
        b1.q[1] = *(const uint4*)(kr + 48);
        v8f c = {};
        c = wmma_f16(aq0.v, b0.v, c);
        c = wmma_f16(aq1.v, b1.v, c);
        for (int r = 0; r < 8; ++r)
            sp[(r + 8 * hi) * SPLD + key0 + ln] = (_Float16)(c[r] * inv_sqrt_hd);
    }
    __syncthreads();

    // ---- phase 2: masked softmax per row ----
    for (int r = 0; r < 16; ++r) {
        _Float16* row = sp + r * SPLD;
        float lmax = -3.0e38f;
        for (int k = tid; k < L_SEQ; k += 128) {
            float v = (float)row[k] + (1.f - maskf[k]) * NEGV;
            lmax = fmaxf(lmax, v);
        }
        const float m = block_reduce_max(lmax, red, tid);
        float lsum = 0.f;
        for (int k = tid; k < L_SEQ; k += 128) {
            float v = (float)row[k] + (1.f - maskf[k]) * NEGV;
            float e = __expf(v - m);
            row[k] = (_Float16)e;
            lsum += e;
        }
        const float is = 1.f / block_reduce_sum(lsum, red, tid);
        for (int k = tid; k < L_SEQ; k += 128)
            row[k] = (_Float16)((float)row[k] * is);
        __syncthreads();
    }

    // ---- phase 3: 3 diffusion steps (causal 5-tap smooth, mask, renorm, blend) ----
    for (int step = 0; step < DIFFST; ++step) {
        for (int r = 0; r < 16; ++r) {
            _Float16* row = sp + r * SPLD;
            float psv[12];
            float lsum = 0.f;
            for (int j = 0; j < 12; ++j) {
                int k = tid + 128 * j;
                float acc = 0.f;
                for (int t = 0; t < KSIZE; ++t) {        // out[k] = sum_t w[t]*p[k-4+t]
                    int kk = k - (KSIZE - 1) + t;
                    if (kk >= 0) acc += wk[t] * (float)row[kk];
                }
                acc *= maskf[k];
                psv[j] = acc;
                lsum += acc;
            }
            const float inv = 1.f / (block_reduce_sum(lsum, red, tid) + EPSV);
            for (int j = 0; j < 12; ++j) {
                int k = tid + 128 * j;
                row[k] = (_Float16)((1.f - ALPHA) * (float)row[k] + ALPHA * psv[j] * inv);
            }
            __syncthreads();
        }
    }

    // ---- phase 4: attn = p @ V  (A from LDS, B from transposed V) ----
    const int d0 = wave * 16;
    const _Float16* vtb = Vt + ((size_t)(b * NHEAD + h) * HDIM + d0 + ln) * L_SEQ;
    v8f acc = {};
    for (int ks = 0; ks < L_SEQ / 32; ++ks) {
        const int key0 = ks * 32;
        F16Frag a;
        const _Float16* prow = sp + ln * SPLD + key0;
        a.q[0] = *(const uint4*)(prow + 8 * hi);            // ds_load_b128
        a.q[1] = *(const uint4*)(prow + 16 + 8 * hi);
        F16Frag bf;
        bf.q[0] = *(const uint4*)(vtb + key0 + 8 * hi);
        bf.q[1] = *(const uint4*)(vtb + key0 + 16 + 8 * hi);
        acc = wmma_f16(a.v, bf.v, acc);
    }
    _Float16* ob = attn + (size_t)(b * L_SEQ + q0) * PDIM + h * HDIM + d0;
    for (int r = 0; r < 8; ++r) {
        int m = r + 8 * hi;
        ob[(size_t)m * PDIM + ln] = (_Float16)acc[r];
    }
}

// ---------------- host launcher ----------------
extern "C" void kernel_launch(void* const* d_in, const int* in_sizes, int n_in,
                              void* d_out, int out_size, void* d_ws, size_t ws_size,
                              hipStream_t stream) {
    const float* hs = (const float*)d_in[0];
    const int*   am = (const int*)d_in[1];
    const float* Wq = (const float*)d_in[2];
    const float* bq = (const float*)d_in[3];
    const float* Wk = (const float*)d_in[4];
    const float* bk = (const float*)d_in[5];
    const float* Wv = (const float*)d_in[6];
    const float* bv = (const float*)d_in[7];
    const float* Wo = (const float*)d_in[8];
    const float* bo = (const float*)d_in[9];
    const float* kw = (const float*)d_in[10];

    // workspace carve (256B aligned)
    char* ws = (char*)d_ws;
    size_t off = 0;
    auto carve = [&](size_t bytes) -> void* {
        void* p = ws + off;
        off = (off + bytes + 255) & ~(size_t)255;
        return p;
    };
    const size_t hsN = (size_t)MROWS * DMODEL;          // 3072*1024
    const size_t wN  = (size_t)DMODEL * PDIM;           // 1024*1024
    _Float16* hs16 = (_Float16*)carve(hsN * 2);
    _Float16* Wqt  = (_Float16*)carve(wN * 2);
    _Float16* Wkt  = (_Float16*)carve(wN * 2);
    _Float16* Wvt  = (_Float16*)carve(wN * 2);
    _Float16* Wot  = (_Float16*)carve(wN * 2);
    _Float16* Q16  = (_Float16*)carve(hsN * 2);
    _Float16* K16  = (_Float16*)carve(hsN * 2);
    _Float16* V16  = (_Float16*)carve(hsN * 2);
    _Float16* Vt16 = (_Float16*)carve(hsN * 2);
    _Float16* At16 = (_Float16*)carve(hsN * 2);
    (void)ws_size;

    // 1) convert activations to f16
    k_cvt_f16<<<(int)((hsN + 255) / 256), 256, 0, stream>>>(hs, hs16, (int)hsN);

    // 2) convert + transpose weights (f32 [K][N] -> f16 [N][K])
    dim3 tgrid(PDIM / 32, DMODEL / 8);
    k_cvt_transpose<<<tgrid, 256, 0, stream>>>(Wq, Wqt, DMODEL, PDIM);
    k_cvt_transpose<<<tgrid, 256, 0, stream>>>(Wk, Wkt, DMODEL, PDIM);
    k_cvt_transpose<<<tgrid, 256, 0, stream>>>(Wv, Wvt, DMODEL, PDIM);
    k_cvt_transpose<<<tgrid, 256, 0, stream>>>(Wo, Wot, DMODEL, PDIM);

    // 3) Q/K/V projections (WMMA GEMM, f16 out + bias), block tile 32x128
    dim3 ggrid(MROWS / 32, PDIM / 128);
    k_gemm<<<ggrid, 128, 0, stream>>>(hs16, DMODEL, Wqt, DMODEL, bq, Q16, PDIM, DMODEL, 0);
    k_gemm<<<ggrid, 128, 0, stream>>>(hs16, DMODEL, Wkt, DMODEL, bk, K16, PDIM, DMODEL, 0);
    k_gemm<<<ggrid, 128, 0, stream>>>(hs16, DMODEL, Wvt, DMODEL, bv, V16, PDIM, DMODEL, 0);

    // 4) transpose V for contiguous p@V B-fragments
    k_transpose_v<<<dim3(L_SEQ / 128, HDIM, BATCH * NHEAD), 128, 0, stream>>>(V16, Vt16);

    // 5) fused attention (scores + softmax + diffusion + p@V) — p never leaves LDS
    k_attention<<<dim3(L_SEQ / 16, NHEAD, BATCH), 128, 0, stream>>>(
        Q16, K16, Vt16, am, kw, At16);

    // 6) output projection (WMMA GEMM, f32 out + bias) -> d_out
    k_gemm<<<ggrid, 128, 0, stream>>>(At16, PDIM, Wot, DMODEL, bo, d_out, DMODEL, DMODEL, 1);
}
